// KARNNLayer_13615046328391
// MI455X (gfx1250) — matrix operand order repaired and checked by reading, not compiled, not run on hardware
//
#include <hip/hip_runtime.h>
#include <hip/hip_bf16.h>

// ---------------- problem constants (from reference) ----------------
#define NB      8                    // NUM + K spline bases
#define NF      9                    // features per scalar input (silu + 8 bases)
#define SEQ     8192
#define IN_DIM  128
#define HID     512
#define OUT_DIM 128
#define HU_IN   (IN_DIM + HID)       // 640
#define HU_F    (HU_IN * NF)         // 5760 folded K for recurrent matvec
#define NWG_REC 64                   // persistent WGs (one per WGP), 8 outputs each
#define OUT_PER_WG (HID / NWG_REC)   // 8
#define KPAD    16                   // padded features/channel for WMMA K tiling
#define KTOT    (HID * KPAD)         // 8192
#define NKT     (KTOT / 32)          // 256 K-chunks of 32
#define M_TILE  64

typedef __attribute__((ext_vector_type(16))) __bf16 v16bf;
typedef __attribute__((ext_vector_type(8)))  float  v8f;

// ---------------- helpers ----------------
__device__ __forceinline__ unsigned short f2bf(float f) {
  unsigned u = __float_as_uint(f);
  u += 0x7fffu + ((u >> 16) & 1u);   // round-to-nearest-even
  return (unsigned short)(u >> 16);
}

// Load the 12 knots + hoist reciprocal knot-differences (loop-invariant).
__device__ __forceinline__ void load_knots(const float* __restrict__ grid,
                                           float* g, float inv[3][11]) {
#pragma unroll
  for (int t = 0; t < 12; ++t) g[t] = grid[t];
#pragma unroll
  for (int d = 1; d <= 3; ++d)
#pragma unroll
    for (int t = 0; t + d < 12; ++t)
      inv[d - 1][t] = 1.0f / (g[t + d] - g[t]);
}

// Cox-de Boor K=3 with precomputed reciprocals (no divides on hot path) + silu.
__device__ __forceinline__ void kan_feat9(float xv, const float* g,
                                          const float inv[3][11], float* o9) {
  float B[11];
#pragma unroll
  for (int t = 0; t < 11; ++t)
    B[t] = (xv >= g[t] && xv < g[t + 1]) ? 1.0f : 0.0f;
#pragma unroll
  for (int d = 1; d <= 3; ++d) {
#pragma unroll
    for (int t = 0; t + d < 11; ++t) {
      float left  = (xv - g[t]) * inv[d - 1][t];
      float right = (g[t + d + 1] - xv) * inv[d - 1][t + 1];
      B[t] = left * B[t] + right * B[t + 1];
    }
  }
  o9[0] = xv * __builtin_amdgcn_rcpf(1.0f + __expf(-xv));   // silu, v_rcp_f32
#pragma unroll
  for (int k = 0; k < 8; ++k) o9[1 + k] = B[k];
}

// ---------------- prep kernels (run every call; ws may be poisoned) ----------------
// Fold recurrent weights: Whu[o][f], f = i*9+j ; j==0 -> scale_base, else ssp*coef
__global__ void fold_hu_kernel(const float* __restrict__ sb, const float* __restrict__ ssp,
                               const float* __restrict__ coef, float* __restrict__ Whu) {
  int idx = blockIdx.x * blockDim.x + threadIdx.x;
  if (idx >= HID * HU_F) return;
  int o = idx / HU_F;
  int f = idx - o * HU_F;
  int i = f / NF;
  int j = f - i * NF;
  int io = i * HID + o;
  float v = (j == 0) ? sb[io] : ssp[io] * coef[io * NB + (j - 1)];
  Whu[idx] = v;
}

// Pack output-layer weights in exact bf16 B-fragment order:
//   slot(((kt*8+nt)*32+lane)*16+e) holds B[K][n], K = kt*32 + (e>=8?16:0) + (lane>=16?8:0) + (e&7)
//   K = c*16 + j ; j==0 base weight, j=1..8 spline weight, j>8 zero pad
__global__ void pack_out_kernel(const float* __restrict__ osb, const float* __restrict__ ossp,
                                const float* __restrict__ ocoef, unsigned short* __restrict__ Bp) {
  int idx = blockIdx.x * blockDim.x + threadIdx.x;
  if (idx >= NKT * 8 * 32 * 16) return;
  int e    = idx & 15;
  int lane = (idx >> 4) & 31;
  int nt   = (idx >> 9) & 7;
  int kt   = idx >> 12;
  int Kk = kt * 32 + ((e & 8) << 1) + ((lane >= 16) ? 8 : 0) + (e & 7);
  int n  = nt * 16 + (lane & 15);
  int c  = Kk >> 4;
  int j  = Kk & 15;
  int cn = c * OUT_DIM + n;
  float v = 0.0f;
  if (j == 0)       v = osb[cn];
  else if (j <= NB) v = ossp[cn] * ocoef[cn * NB + (j - 1)];
  Bp[idx] = f2bf(v);
}

__global__ void init_kernel(const float* __restrict__ h0, float* hbufs, unsigned* bar) {
  int t = threadIdx.x;
  if (t < HID) { hbufs[t] = h0[t]; hbufs[HID + t] = 0.0f; }
  if (t == 0)  { bar[0] = 0u; bar[1] = 0u; }
}

// ---------------- persistent recurrence: weights pinned in 320KB LDS ----------------
__global__ void __launch_bounds__(256, 1) recurrence_kernel(
    const float* __restrict__ x, const float* __restrict__ grid,
    const float* __restrict__ Whu, float* hbufs,
    float* __restrict__ hidden_out, unsigned* bar) {
  extern __shared__ float smem[];
  float* wlds = smem;                        // [8][5760] fp32 = 184KB
  float* feat = smem + OUT_PER_WG * HU_F;    // [5760]    = 23KB
  const int tid  = threadIdx.x;
  const int wg   = blockIdx.x;
  const int wv   = tid >> 5;
  const int lane = tid & 31;

  float g[12], inv[3][11];
  load_knots(grid, g, inv);

  // Pin this WG's 8-output weight slice into LDS once.
  for (int o = 0; o < OUT_PER_WG; ++o) {
    const float* src = Whu + (size_t)(wg * OUT_PER_WG + o) * HU_F;
    for (int f = tid; f < HU_F; f += 256) wlds[o * HU_F + f] = src[f];
  }
  __syncthreads();

  const float4* feat4 = (const float4*)feat;
  const float4* w4    = (const float4*)(wlds + wv * HU_F);

  unsigned gen = 0;
  for (int t = 0; t < SEQ; ++t) {
    const float* hprev = hbufs + (t & 1) * HID;
    float*       hnext = hbufs + ((t + 1) & 1) * HID;

    // features for inp = [x_t ; h_prev] (computed redundantly per WG)
    for (int i = tid; i < HU_IN; i += 256) {
      float v = (i < IN_DIM)
          ? x[(size_t)t * IN_DIM + i]
          : __hip_atomic_load(&hprev[i - IN_DIM], __ATOMIC_RELAXED, __HIP_MEMORY_SCOPE_AGENT);
      float f9[NF];
      kan_feat9(v, g, inv, f9);
#pragma unroll
      for (int j = 0; j < NF; ++j) feat[i * NF + j] = f9[j];
    }
    __syncthreads();

    // one wave per output: float4 LDS matvec (2x ds_load_b128 + 4 FMA per iter)
    float acc = 0.0f;
    for (int f4 = lane; f4 < HU_F / 4; f4 += 32) {
      float4 fv = feat4[f4];
      float4 wv4 = w4[f4];
      acc += fv.x * wv4.x + fv.y * wv4.y + fv.z * wv4.z + fv.w * wv4.w;
    }
#pragma unroll
    for (int off = 16; off > 0; off >>= 1) acc += __shfl_down(acc, off, 32);
    if (lane == 0) {
      int o = wg * OUT_PER_WG + wv;
      hidden_out[(size_t)t * HID + o] = acc;
      __hip_atomic_store(&hnext[o], acc, __ATOMIC_RELAXED, __HIP_MEMORY_SCOPE_AGENT);
    }

    // device-wide generation barrier (one per step)
    __syncthreads();
    if (tid == 0) {
      __threadfence();
      unsigned prev = atomicAdd(&bar[0], 1u);
      if (prev == NWG_REC - 1) {
        atomicExch(&bar[0], 0u);
        __threadfence();
        atomicAdd(&bar[1], 1u);
      } else {
        while (__hip_atomic_load(&bar[1], __ATOMIC_RELAXED, __HIP_MEMORY_SCOPE_AGENT) <= gen)
          __builtin_amdgcn_s_sleep(1);
      }
      __threadfence();
    }
    gen++;
    __syncthreads();
  }
}

// ---------------- output layer: fused feature-gen + bf16 WMMA GEMM ----------------
// D(8192x128) = A(8192 x 8192pad) * B(8192pad x 128), A built on the fly in LDS.
// Per outer iteration: stage 64 K (4 channels) with ALL 256 threads, then 8 WMMAs.
__global__ void __launch_bounds__(256, 2) out_gemm_kernel(
    const float* __restrict__ hidden, const float* __restrict__ grid,
    const unsigned short* __restrict__ Bp, float* __restrict__ out) {
  __shared__ __align__(32) unsigned short A_sh[2][4 * 32 * 16];  // 2 sub-chunks x 4 M-tiles
  const int tid  = threadIdx.x;
  const int wv   = tid >> 5;       // wave -> N-tile (8 waves x 16 cols = 128)
  const int lane = tid & 31;
  const int m0   = blockIdx.x * M_TILE;

  float g[12], inv[3][11];
  load_knots(grid, g, inv);

  v8f acc[4] = {};

  for (int ktp = 0; ktp < NKT / 2; ++ktp) {     // 128 outer iters, 64 K each
    // Stage A: 64 rows x 4 channels -> bf16 fragments in ISA layout (all threads)
    {
      int r   = tid & 63;
      int cs2 = tid >> 6;            // 0..3: channel within 64-K chunk
      int sub = cs2 >> 1;            // which 32-K sub-chunk
      int cs  = cs2 & 1;             // channel within sub-chunk
      int c   = ktp * 4 + cs2;
      float h = hidden[(size_t)(m0 + r) * HID + c];
      float f9[NF];
      kan_feat9(h, g, inv, f9);
      union { unsigned short us[8]; uint4 v; } lo, hi;
#pragma unroll
      for (int j = 0; j < 8; ++j) lo.us[j] = f2bf(f9[j]);      // j = 0..7
      hi.us[0] = f2bf(f9[8]);                                   // j = 8
#pragma unroll
      for (int j = 1; j < 8; ++j) hi.us[j] = 0;                 // j = 9..15 pad
      int mt = r >> 4, ri = r & 15;
      *(uint4*)&A_sh[sub][(mt * 32 + ri)      * 16 + cs * 8] = lo.v; // lanes 0-15
      *(uint4*)&A_sh[sub][(mt * 32 + ri + 16) * 16 + cs * 8] = hi.v; // lanes 16-31
    }
    __syncthreads();

#pragma unroll
    for (int sub = 0; sub < 2; ++sub) {
      int kt = ktp * 2 + sub;
      // B fragment: one contiguous 32B load per lane (pre-packed)
      v16bf bfrag = *(const v16bf*)&Bp[(size_t)(((kt * 8 + wv) * 32 + lane) * 16)];
#pragma unroll
      for (int mt = 0; mt < 4; ++mt) {
        v16bf afrag = *(const v16bf*)&A_sh[sub][(mt * 32 + lane) * 16];
        acc[mt] = __builtin_amdgcn_wmma_f32_16x16x32_bf16(
            false, afrag, false, bfrag, (short)0, acc[mt], false, false);
      }
    }
    __syncthreads();
  }

  // C/D layout: VGPR r -> row mt*16 + r (+8 for lanes 16-31), col = lane&15
#pragma unroll
  for (int mt = 0; mt < 4; ++mt) {
#pragma unroll
    for (int r = 0; r < 8; ++r) {
      int row = m0 + mt * 16 + r + ((lane >= 16) ? 8 : 0);
      int col = wv * 16 + (lane & 15);
      out[(size_t)row * OUT_DIM + col] = acc[mt][r];
    }
  }
}

// ---------------- host ----------------
extern "C" void kernel_launch(void* const* d_in, const int* in_sizes, int n_in,
                              void* d_out, int out_size, void* d_ws, size_t ws_size,
                              hipStream_t stream) {
  const float* x    = (const float*)d_in[0];
  const float* h0   = (const float*)d_in[1];
  const float* hug  = (const float*)d_in[2];
  const float* huc  = (const float*)d_in[3];
  const float* husb = (const float*)d_in[4];
  const float* husp = (const float*)d_in[5];
  const float* og   = (const float*)d_in[6];
  const float* oc   = (const float*)d_in[7];
  const float* osb  = (const float*)d_in[8];
  const float* osp  = (const float*)d_in[9];

  float* out        = (float*)d_out;                 // (8192,128)
  float* hidden_out = out + (size_t)SEQ * OUT_DIM;   // (8192,512) second tuple output

  // workspace layout (~14 MB)
  char* ws = (char*)d_ws;
  float*          Whu   = (float*)ws;                                            // 11.8 MB
  unsigned short* Bp    = (unsigned short*)(ws + (size_t)HID * HU_F * 4);        //  2 MB
  float*          hbufs = (float*)((char*)Bp + (size_t)KTOT * OUT_DIM * 2);      //  4 KB
  unsigned*       bar   = (unsigned*)(hbufs + 2 * HID);

  {
    int total = HID * HU_F;
    fold_hu_kernel<<<(total + 255) / 256, 256, 0, stream>>>(husb, husp, huc, Whu);
  }
  {
    int total = NKT * 8 * 32 * 16;
    pack_out_kernel<<<(total + 255) / 256, 256, 0, stream>>>(osb, osp, oc, Bp);
  }
  init_kernel<<<1, 512, 0, stream>>>(h0, hbufs, bar);

  size_t smem = (size_t)(OUT_PER_WG * HU_F + HU_F) * sizeof(float);  // ~207 KB of WGP LDS
  hipFuncSetAttribute((const void*)recurrence_kernel,
                      hipFuncAttributeMaxDynamicSharedMemorySize, (int)smem);
  recurrence_kernel<<<NWG_REC, 256, smem, stream>>>(x, hug, Whu, hbufs, hidden_out, bar);

  out_gemm_kernel<<<SEQ / M_TILE, 256, 0, stream>>>(hidden_out, og, Bp, out);
}
